// GPT_49237505081880
// MI455X (gfx1250) — compile-verified, matrix operand order checked by reading, not run on hardware
//
#include <hip/hip_runtime.h>
#include <hip/hip_bf16.h>
#include <math.h>

// ---------------- problem constants (match reference) ----------------
enum { L_ = 2, H_ = 12, N_ = 64, C_ = 768, V_ = 50304, S_ = 1024,
       B_ = 2, T_ = 1024, Q_ = 256, FF_ = 2688, DMIX = 32, DDEC = 64,
       NC = T_ / Q_, BT = B_ * T_, BH = B_ * H_ };

typedef __attribute__((ext_vector_type(16))) __bf16 v16bf;
typedef __attribute__((ext_vector_type(8)))  __bf16 v8bf;
typedef __attribute__((ext_vector_type(8)))  float  v8f;

// =====================================================================
// WMMA bf16 GEMM:  Out[M,N] = act(A[M,K] @ B + bias) (+ resid)
//   TRANSB=0: B row-major [K,N].  TRANSB=1: B row-major [N,K] (uses B^T).
// Block = 256 thr = 8 waves. Block tile: 128(M) x 64(N). Wave -> 16x64.
// A(128x32) and B(64x32, stored [n][k]) staged in LDS as bf16; each WMMA
// fragment is two 16B ds_load_b128 (K offsets are two contiguous 8-runs:
// k = g*8 + e  and  k = 16 + g*8 + e, per CDNA5 ISA 7.12.2).
// C/D f32: VGPR i -> row i + 8*(lane>>4); lane&15 -> column.
// =====================================================================
template <int TRANSB>
__global__ void k_gemm_t(const float* __restrict__ A, const float* __restrict__ Bm,
                         const float* __restrict__ bias, const float* __restrict__ resid,
                         float* __restrict__ Out, int M, int N, int K, int act) {
  __shared__ __attribute__((aligned(16))) __bf16 lsA[128 * 32];
  __shared__ __attribute__((aligned(16))) __bf16 lsB[64 * 32];
  const int tid  = threadIdx.x;
  const int lane = tid & 31;
  const int wave = tid >> 5;
  const int g    = lane >> 4;        // K-half select
  const int rc   = lane & 15;        // row (A) / col (B,D)
  const int mBlk = blockIdx.y * 128;
  const int n0   = blockIdx.x * 64;

  // cooperative-loader indices (all compile-time strides)
  const int arow = tid >> 1, akh = (tid & 1) * 16;       // A: 128 rows x 2 halves of 16
  const int tbn  = tid >> 2, tbkh = (tid & 3) * 8;       // B trans: 64 rows x 4 halves of 8
  const int bkrow = tid >> 3, bng = (tid & 7) * 8;       // B non-trans: 32 k-rows x 8 n-groups

  v8f acc[4] = {};
  for (int k0 = 0; k0 < K; k0 += 32) {
    // ---- stage A tile (coalesced, contiguous 16 floats per thread) ----
    {
      const float* ap = A + (size_t)(mBlk + arow) * K + k0 + akh;
      __builtin_prefetch(ap + 32, 0, 1);
#pragma unroll
      for (int j = 0; j < 16; ++j) lsA[arow * 32 + akh + j] = (__bf16)ap[j];
    }
    // ---- stage B tile into [n][k] layout ----
    if (TRANSB) {
      const float* bp = Bm + (size_t)(n0 + tbn) * K + k0 + tbkh;
      __builtin_prefetch(bp + 32, 0, 1);
#pragma unroll
      for (int j = 0; j < 8; ++j) lsB[tbn * 32 + tbkh + j] = (__bf16)bp[j];
    } else {
      const float* bp = Bm + (size_t)(k0 + bkrow) * N + n0 + bng;
#pragma unroll
      for (int j = 0; j < 8; ++j) lsB[(bng + j) * 32 + bkrow] = (__bf16)bp[j];
    }
    __syncthreads();
    // ---- fragments from LDS (2x ds_load_b128 each) + WMMA ----
    const int abase = (wave * 16 + rc) * 32 + g * 8;
    const v8bf alo = *(const v8bf*)&lsA[abase];
    const v8bf ahi = *(const v8bf*)&lsA[abase + 16];
    const v16bf a = __builtin_shufflevector(alo, ahi, 0, 1, 2, 3, 4, 5, 6, 7,
                                            8, 9, 10, 11, 12, 13, 14, 15);
#pragma unroll
    for (int nt = 0; nt < 4; ++nt) {
      const int bbase = (nt * 16 + rc) * 32 + g * 8;
      const v8bf blo = *(const v8bf*)&lsB[bbase];
      const v8bf bhi = *(const v8bf*)&lsB[bbase + 16];
      const v16bf b = __builtin_shufflevector(blo, bhi, 0, 1, 2, 3, 4, 5, 6, 7,
                                              8, 9, 10, 11, 12, 13, 14, 15);
      acc[nt] = __builtin_amdgcn_wmma_f32_16x16x32_bf16(
          false, a, false, b, (short)0, acc[nt], false, false);
    }
    __syncthreads();
  }
  // ---- epilogue ----
  const int m0 = mBlk + wave * 16;
#pragma unroll
  for (int nt = 0; nt < 4; ++nt) {
    const int n = n0 + nt * 16 + rc;
    const float bv = bias ? bias[n] : 0.f;
#pragma unroll
    for (int i = 0; i < 8; ++i) {
      const int m = m0 + g * 8 + i;
      float v = acc[nt][i] + bv;
      if (act == 1) { v = fmaxf(v, 0.f); v *= v; }          // relu^2
      else if (act == 2) { v = 1.f / (1.f + expf(-v)); }    // sigmoid
      if (resid) v += resid[(size_t)m * N + n];
      Out[(size_t)m * N + n] = v;
    }
  }
}

// ---------------- embedding ----------------
__global__ void k_embed(const int* __restrict__ idx, const float* __restrict__ wte,
                        const float* __restrict__ wpe, float* __restrict__ x) {
  const int tok = blockIdx.x;
  const int t = tok % T_;
  const int id = idx[tok];
  for (int c = threadIdx.x; c < C_; c += 256)
    x[(size_t)tok * C_ + c] = wte[(size_t)id * C_ + c] + wpe[(size_t)t * C_ + c];
}

// ---------------- LayerNorm over C (one token per block) ----------------
__global__ void k_ln(const float* __restrict__ in, const float* __restrict__ w,
                     const float* __restrict__ b, float* __restrict__ out) {
  const int tok = blockIdx.x;
  const float* xr = in + (size_t)tok * C_;
  __shared__ float red[256];
  float s = 0.f;
  for (int c = threadIdx.x; c < C_; c += 256) s += xr[c];
  red[threadIdx.x] = s; __syncthreads();
  for (int o = 128; o > 0; o >>= 1) { if (threadIdx.x < o) red[threadIdx.x] += red[threadIdx.x + o]; __syncthreads(); }
  const float mean = red[0] / (float)C_;
  __syncthreads();
  float s2 = 0.f;
  for (int c = threadIdx.x; c < C_; c += 256) { float d = xr[c] - mean; s2 += d * d; }
  red[threadIdx.x] = s2; __syncthreads();
  for (int o = 128; o > 0; o >>= 1) { if (threadIdx.x < o) red[threadIdx.x] += red[threadIdx.x + o]; __syncthreads(); }
  const float rstd = rsqrtf(red[0] / (float)C_ + 1e-5f);
  for (int c = threadIdx.x; c < C_; c += 256)
    out[(size_t)tok * C_ + c] = (xr[c] - mean) * rstd * w[c] + b[c];
}

// ---------------- TimeMix LoRA token-shift mixing ----------------
__global__ void k_lora(const float* __restrict__ xln,
                       const float* __restrict__ maax, const float* __restrict__ maaw,
                       const float* __restrict__ maak, const float* __restrict__ maav,
                       const float* __restrict__ maar,
                       const float* __restrict__ w1, const float* __restrict__ w2,
                       float* __restrict__ xw, float* __restrict__ xk,
                       float* __restrict__ xv, float* __restrict__ xr) {
  const int tok = blockIdx.x;
  const int t = tok % T_;
  __shared__ float sx[C_], sxx[C_], sxm[C_], stv[4 * DMIX];
  const float* xc = xln + (size_t)tok * C_;
  for (int c = threadIdx.x; c < C_; c += 256) {
    const float cur = xc[c];
    const float prev = (t > 0) ? xc[c - C_] : 0.f;
    const float dx = prev - cur;
    sx[c] = cur; sxx[c] = dx; sxm[c] = cur + dx * maax[c];
  }
  __syncthreads();
  if (threadIdx.x < 4 * DMIX) {
    float a = 0.f;
    for (int c = 0; c < C_; ++c) a += sxm[c] * w1[(size_t)c * (4 * DMIX) + threadIdx.x];
    stv[threadIdx.x] = tanhf(a);
  }
  __syncthreads();
  for (int c = threadIdx.x; c < C_; c += 256) {
    float m[4];
#pragma unroll
    for (int f = 0; f < 4; ++f) {
      float a = 0.f;
      const float* w2f = w2 + (size_t)f * DMIX * C_ + c;
      for (int r = 0; r < DMIX; ++r) a += stv[f * DMIX + r] * w2f[(size_t)r * C_];
      m[f] = a;
    }
    const float cur = sx[c], dx = sxx[c];
    const size_t o = (size_t)tok * C_ + c;
    xw[o] = cur + dx * (maaw[c] + m[0]);
    xk[o] = cur + dx * (maak[c] + m[1]);
    xv[o] = cur + dx * (maav[c] + m[2]);
    xr[o] = cur + dx * (maar[c] + m[3]);
  }
}

// ---------------- data-dependent decay -> per-head wlog ----------------
__global__ void k_decay(const float* __restrict__ xw, const float* __restrict__ dw1,
                        const float* __restrict__ dw2, const float* __restrict__ td,
                        float* __restrict__ wlog) {
  const int tok = blockIdx.x;
  const int b = tok / T_, t = tok % T_;
  __shared__ float sx[C_], sh[DDEC], sw[C_];
  for (int c = threadIdx.x; c < C_; c += 256) sx[c] = xw[(size_t)tok * C_ + c];
  __syncthreads();
  if (threadIdx.x < DDEC) {
    float a = 0.f;
    for (int c = 0; c < C_; ++c) a += sx[c] * dw1[(size_t)c * DDEC + threadIdx.x];
    sh[threadIdx.x] = tanhf(a);
  }
  __syncthreads();
  for (int c = threadIdx.x; c < C_; c += 256) {
    float a = td[c];
    for (int d = 0; d < DDEC; ++d) a += sh[d] * dw2[(size_t)d * C_ + c];
    sw[c] = a;
  }
  __syncthreads();
  if (threadIdx.x < H_) {
    float a = 0.f;
    for (int n = 0; n < N_; ++n) a += sw[threadIdx.x * N_ + n];
    a *= (1.f / (float)N_);
    wlog[((size_t)(b * H_ + threadIdx.x)) * T_ + t] = -expf(a);
  }
}

// ---------------- per-chunk inclusive cumsum of wlog ----------------
__global__ void k_cumsum(const float* __restrict__ wlog, float* __restrict__ cum,
                         float* __restrict__ wsum) {
  const int blk = blockIdx.x;          // bh*NC + chunk
  const int bh = blk / NC, c = blk % NC;
  const int q = threadIdx.x;
  __shared__ float s[Q_];
  s[q] = wlog[(size_t)bh * T_ + c * Q_ + q];
  __syncthreads();
  for (int off = 1; off < Q_; off <<= 1) {
    const float add = (q >= off) ? s[q - off] : 0.f;
    __syncthreads();
    s[q] += add;
    __syncthreads();
  }
  cum[(size_t)bh * T_ + c * Q_ + q] = s[q];
  if (q == Q_ - 1) wsum[blk] = s[q];
}

// ---------------- per-chunk kv = (k * w_inter)^T @ v  (64x64) ----------------
__global__ void k_kv(const float* __restrict__ kb, const float* __restrict__ vb,
                     const float* __restrict__ cum, const float* __restrict__ wsum,
                     float* __restrict__ kv) {
  const int blk = blockIdx.x;
  const int bh = blk / NC, c = blk % NC;
  const int b = bh / H_, h = bh % H_;
  __shared__ float wi[Q_];
  wi[threadIdx.x] = expf(wsum[blk] - cum[(size_t)bh * T_ + c * Q_ + threadIdx.x]);
  __syncthreads();
  const int n = threadIdx.x & 63, mg = threadIdx.x >> 6;
  float acc[16];
#pragma unroll
  for (int i = 0; i < 16; ++i) acc[i] = 0.f;
  for (int q = 0; q < Q_; ++q) {
    const size_t base = ((size_t)(b * T_ + c * Q_ + q)) * C_ + h * N_;
    const float kq = kb[base + n] * wi[q];
    const float* vp = vb + base + mg * 16;
#pragma unroll
    for (int i = 0; i < 16; ++i) acc[i] += kq * vp[i];
  }
  float* op = kv + ((size_t)blk * N_ + n) * N_ + mg * 16;
#pragma unroll
  for (int i = 0; i < 16; ++i) op[i] = acc[i];
}

// ---------------- sequential chunk-state scan ----------------
__global__ void k_states(const float* __restrict__ kv, const float* __restrict__ wsum,
                         float* __restrict__ st) {
  const int bh = blockIdx.x;
  const int tid = threadIdx.x;
  float s[16];
#pragma unroll
  for (int i = 0; i < 16; ++i) s[i] = 0.f;
  for (int c = 0; c < NC; ++c) {
    const size_t base = ((size_t)(bh * NC + c)) * (N_ * N_) + (size_t)tid * 16;
#pragma unroll
    for (int i = 0; i < 16; ++i) st[base + i] = s[i];        // state at chunk start
    const float dec = expf(wsum[bh * NC + c]);
#pragma unroll
    for (int i = 0; i < 16; ++i) s[i] = dec * s[i] + kv[base + i];
  }
}

// ---------------- intra-chunk attention + state readout ----------------
__global__ void k_yout(const float* __restrict__ rb, const float* __restrict__ kb,
                       const float* __restrict__ vb, const float* __restrict__ cum,
                       const float* __restrict__ wlog, const float* __restrict__ st,
                       const float* __restrict__ faaaa, float* __restrict__ yb) {
  const int blk = blockIdx.x, mg = blockIdx.y;
  const int bh = blk / NC, c = blk % NC;
  const int b = bh / H_, h = bh % H_;
  const int q = threadIdx.x;
  const int t = c * Q_ + q;
  __shared__ float cs[Q_];
  cs[q] = cum[(size_t)bh * T_ + t];
  __syncthreads();
  const size_t rowq = ((size_t)(b * T_ + t)) * C_ + h * N_;
  float r[N_];
#pragma unroll
  for (int n = 0; n < N_; ++n) r[n] = rb[rowq + n];
  const float sh = cs[q] - wlog[(size_t)bh * T_ + t];   // cum - wc (== shifted)
  const float wintra = expf(sh);
  float acc[16];
#pragma unroll
  for (int i = 0; i < 16; ++i) acc[i] = 0.f;
  // diagonal (faaaa) term
  float dq = 0.f;
  const float* fa = faaaa + h * N_;
#pragma unroll
  for (int n = 0; n < N_; ++n) dq += r[n] * kb[rowq + n] * fa[n];
  const float* vq = vb + rowq + mg * 16;
#pragma unroll
  for (int i = 0; i < 16; ++i) acc[i] += dq * vq[i];
  // inter-chunk: (r * w_intra) @ state
  const float* stc = st + ((size_t)(bh * NC + c)) * (N_ * N_);
#pragma unroll
  for (int n = 0; n < N_; ++n) {
    const float rw = r[n] * wintra;
    const float* sp = stc + n * N_ + mg * 16;
#pragma unroll
    for (int i = 0; i < 16; ++i) acc[i] += rw * sp[i];
  }
  // strictly-causal intra-chunk attention
  for (int s = 0; s < q; ++s) {
    const size_t rows = ((size_t)(b * T_ + c * Q_ + s)) * C_ + h * N_;
    float rk = 0.f;
#pragma unroll
    for (int n = 0; n < N_; ++n) rk += r[n] * kb[rows + n];
    const float a = rk * expf(sh - cs[s]);
    const float* vs = vb + rows + mg * 16;
#pragma unroll
    for (int i = 0; i < 16; ++i) acc[i] += a * vs[i];
  }
  float* yp = yb + rowq + mg * 16;
#pragma unroll
  for (int i = 0; i < 16; ++i) yp[i] = acc[i];
}

// ---------------- ChannelMix token shift ----------------
__global__ void k_cshift(const float* __restrict__ xln, const float* __restrict__ cmk,
                         const float* __restrict__ cmr, float* __restrict__ xk2,
                         float* __restrict__ xr2) {
  const int tok = blockIdx.x;
  const int t = tok % T_;
  const float* xc = xln + (size_t)tok * C_;
  for (int c = threadIdx.x; c < C_; c += 256) {
    const float cur = xc[c];
    const float prev = (t > 0) ? xc[c - C_] : 0.f;
    const float dx = prev - cur;
    xk2[(size_t)tok * C_ + c] = cur + dx * cmk[c];
    xr2[(size_t)tok * C_ + c] = cur + dx * cmr[c];
  }
}

// ---------------- x += sig * hv ----------------
__global__ void k_mulacc(float* __restrict__ x, const float* __restrict__ a,
                         const float* __restrict__ b, int n) {
  const int i = blockIdx.x * 256 + threadIdx.x;
  if (i < n) x[i] += a[i] * b[i];
}

// ---------------- per-token CE loss ----------------
__global__ void k_tokloss(const float* __restrict__ logits, const int* __restrict__ targets,
                          float* __restrict__ tl) {
  const int tok = blockIdx.x;
  const float* lg = logits + (size_t)tok * V_;
  __shared__ float red[256];
  float mx = -1e30f;
  for (int v = threadIdx.x; v < V_; v += 256) mx = fmaxf(mx, lg[v]);
  red[threadIdx.x] = mx; __syncthreads();
  for (int o = 128; o > 0; o >>= 1) { if (threadIdx.x < o) red[threadIdx.x] = fmaxf(red[threadIdx.x], red[threadIdx.x + o]); __syncthreads(); }
  mx = red[0]; __syncthreads();
  float se = 0.f;
  for (int v = threadIdx.x; v < V_; v += 256) se += expf(lg[v] - mx);
  red[threadIdx.x] = se; __syncthreads();
  for (int o = 128; o > 0; o >>= 1) { if (threadIdx.x < o) red[threadIdx.x] += red[threadIdx.x + o]; __syncthreads(); }
  if (threadIdx.x == 0) tl[tok] = logf(red[0]) + mx - lg[targets[tok]];
}

__global__ void k_lossred(const float* __restrict__ tl, float* __restrict__ out) {
  __shared__ float red[256];
  float s = 0.f;
  for (int i = threadIdx.x; i < BT; i += 256) s += tl[i];
  red[threadIdx.x] = s; __syncthreads();
  for (int o = 128; o > 0; o >>= 1) { if (threadIdx.x < o) red[threadIdx.x] += red[threadIdx.x + o]; __syncthreads(); }
  if (threadIdx.x == 0) out[0] = red[0] / (float)BT;
}

// =====================================================================
static inline void gemm(const float* A, const float* Bm, const float* bias,
                        const float* resid, float* Out, int M, int N, int K,
                        int act, int transb, hipStream_t s) {
  dim3 grid(N / 64, M / 128);
  if (transb)
    k_gemm_t<1><<<grid, 256, 0, s>>>(A, Bm, bias, resid, Out, M, N, K, act);
  else
    k_gemm_t<0><<<grid, 256, 0, s>>>(A, Bm, bias, resid, Out, M, N, K, act);
}

extern "C" void kernel_launch(void* const* d_in, const int* in_sizes, int n_in,
                              void* d_out, int out_size, void* d_ws, size_t ws_size,
                              hipStream_t stream) {
  (void)in_sizes; (void)n_in; (void)out_size; (void)ws_size;
  const int*   idx     = (const int*)d_in[0];
  const int*   targets = (const int*)d_in[1];
  const float* wte     = (const float*)d_in[2];
  const float* wpe     = (const float*)d_in[3];
  const float* ln1w    = (const float*)d_in[4];
  const float* ln1b    = (const float*)d_in[5];
  const float* ln2w    = (const float*)d_in[6];
  const float* ln2b    = (const float*)d_in[7];
  const float* maax    = (const float*)d_in[8];
  const float* maaw    = (const float*)d_in[9];
  const float* maak    = (const float*)d_in[10];
  const float* maav    = (const float*)d_in[11];
  const float* maar    = (const float*)d_in[12];
  const float* maaw1   = (const float*)d_in[13];
  const float* maaw2   = (const float*)d_in[14];
  const float* decw1   = (const float*)d_in[15];
  const float* decw2   = (const float*)d_in[16];
  const float* td      = (const float*)d_in[17];
  const float* faaaa   = (const float*)d_in[18];
  const float* Wr      = (const float*)d_in[19];
  const float* br      = (const float*)d_in[20];
  const float* Wk      = (const float*)d_in[21];
  const float* bk      = (const float*)d_in[22];
  const float* Wv      = (const float*)d_in[23];
  const float* bv      = (const float*)d_in[24];
  const float* Wo      = (const float*)d_in[25];
  const float* bo      = (const float*)d_in[26];
  const float* lnxw    = (const float*)d_in[27];
  const float* lnxb    = (const float*)d_in[28];
  const float* cmk     = (const float*)d_in[29];
  const float* cmr     = (const float*)d_in[30];
  const float* cWk     = (const float*)d_in[31];
  const float* cbk     = (const float*)d_in[32];
  const float* cWv     = (const float*)d_in[33];
  const float* cbv     = (const float*)d_in[34];
  const float* cWr     = (const float*)d_in[35];
  const float* cbr     = (const float*)d_in[36];
  const float* lnfw    = (const float*)d_in[37];
  const float* lnfb    = (const float*)d_in[38];

  // workspace carve-up
  float* w = (float*)d_ws;
  auto take = [&](size_t n) { float* p = w; w += n; return p; };
  const size_t TOK = (size_t)BT * C_;
  float* xb   = take(TOK);
  float* xlnb = take(TOK);
  float* xwb  = take(TOK);
  float* xkb  = take(TOK);
  float* xvb  = take(TOK);
  float* xrb  = take(TOK);
  float* rb   = take(TOK);
  float* kb   = take(TOK);
  float* vb   = take(TOK);
  float* yb   = take(TOK);
  float* tmpb = take(TOK);
  float* sgb  = take(TOK);
  float* hb   = take((size_t)BT * FF_);
  float* wlogb = take((size_t)BH * T_);
  float* cumb  = take((size_t)BH * T_);
  float* wsumb = take((size_t)BH * NC);
  float* kvb   = take((size_t)BH * NC * N_ * N_);
  float* stb   = take((size_t)BH * NC * N_ * N_);
  float* tlb   = take((size_t)BT);

  k_embed<<<BT, 256, 0, stream>>>(idx, wte, wpe, xb);

  for (int l = 0; l < L_; ++l) {
    const size_t oC  = (size_t)l * C_;
    const size_t oCC = (size_t)l * C_ * C_;
    // ---- TimeMix ----
    k_ln<<<BT, 256, 0, stream>>>(xb, ln1w + oC, ln1b + oC, xlnb);
    k_lora<<<BT, 256, 0, stream>>>(xlnb, maax + oC, maaw + oC, maak + oC, maav + oC,
                                   maar + oC, maaw1 + (size_t)l * C_ * 4 * DMIX,
                                   maaw2 + (size_t)l * 4 * DMIX * C_,
                                   xwb, xkb, xvb, xrb);
    gemm(xrb, Wr + oCC, br + oC, nullptr, rb, BT, C_, C_, 0, 0, stream);
    gemm(xkb, Wk + oCC, bk + oC, nullptr, kb, BT, C_, C_, 0, 0, stream);
    gemm(xvb, Wv + oCC, bv + oC, nullptr, vb, BT, C_, C_, 0, 0, stream);
    k_decay<<<BT, 256, 0, stream>>>(xwb, decw1 + (size_t)l * C_ * DDEC,
                                    decw2 + (size_t)l * DDEC * C_, td + oC, wlogb);
    k_cumsum<<<BH * NC, Q_, 0, stream>>>(wlogb, cumb, wsumb);
    k_kv<<<BH * NC, 256, 0, stream>>>(kb, vb, cumb, wsumb, kvb);
    k_states<<<BH, 256, 0, stream>>>(kvb, wsumb, stb);
    dim3 yg(BH * NC, 4);
    k_yout<<<yg, Q_, 0, stream>>>(rb, kb, vb, cumb, wlogb, stb,
                                  faaaa + (size_t)l * H_ * N_, yb);
    k_ln<<<BT, 256, 0, stream>>>(yb, lnxw + oC, lnxb + oC, tmpb);
    gemm(tmpb, Wo + oCC, bo + oC, xb, xb, BT, C_, C_, 0, 0, stream);  // + residual
    // ---- ChannelMix ----
    k_ln<<<BT, 256, 0, stream>>>(xb, ln2w + oC, ln2b + oC, xlnb);
    k_cshift<<<BT, 256, 0, stream>>>(xlnb, cmk + oC, cmr + oC, xkb, xrb);
    gemm(xkb, cWk + (size_t)l * C_ * FF_, cbk + (size_t)l * FF_, nullptr, hb,
         BT, FF_, C_, /*relu^2*/1, 0, stream);
    gemm(xrb, cWr + oCC, cbr + oC, nullptr, sgb, BT, C_, C_, /*sigmoid*/2, 0, stream);
    gemm(hb, cWv + (size_t)l * FF_ * C_, cbv + oC, nullptr, tmpb, BT, C_, FF_, 0, 0, stream);
    k_mulacc<<<((int)TOK + 255) / 256, 256, 0, stream>>>(xb, sgb, tmpb, (int)TOK);
  }

  // ---- head ----
  k_ln<<<BT, 256, 0, stream>>>(xb, lnfw, lnfb, xlnb);
  float* logits = (float*)d_out;
  gemm(xlnb, wte, nullptr, nullptr, logits, BT, V_, C_, 0, /*transb*/1, stream);
  k_tokloss<<<BT, 256, 0, stream>>>(logits, targets, tlb);
  k_lossred<<<1, 256, 0, stream>>>(tlb, logits + (size_t)BT * V_);
}
